// FusionAttentionAspectExtractionV2_17076789969509
// MI455X (gfx1250) — matrix operand order repaired and checked by reading, not compiled, not run on hardware
//
#include <hip/hip_runtime.h>
#include <hip/hip_bf16.h>

#define B_ 32
#define S_ 512
#define E_ 300
#define H_ 256
#define H2_ 512
#define H3_ 768
#define H4_ 1024
#define NEG_ 1e10f

// CDNA5 async copy-to-LDS (ASYNCcnt-tracked DMA, no VGPR staging); guarded so
// the file still compiles on toolchains without these builtins. The builtin's
// pointer params are generic int4* (per compiler diagnostic), 16B granules.
#if __has_builtin(__builtin_amdgcn_global_load_async_to_lds_b128) && \
    __has_builtin(__builtin_amdgcn_s_wait_asynccnt)
#define ASYNC_LDS 1
typedef int v4i_async __attribute__((vector_size(16)));
#define ASYNC_PTR(p) ((v4i_async*)(p))
#else
#define ASYNC_LDS 0
#endif

typedef float v2f __attribute__((ext_vector_type(2)));
typedef float v8f __attribute__((ext_vector_type(8)));
typedef _Float16 v8h __attribute__((ext_vector_type(8)));
typedef _Float16 v16h __attribute__((ext_vector_type(16)));

__device__ __forceinline__ v8f wmma4(v2f a, v2f b, v8f c) {
  // D = A(16x4,f32) x B(4x16,f32) + C(16x16,f32)
  return __builtin_amdgcn_wmma_f32_16x16x4_f32(false, a, false, b, (short)0, c,
                                               false, false);
}
__device__ __forceinline__ v8f wmma32h(v16h a, v16h b, v8f c) {
  // D = A(16x32,f16) x B(32x16,f16) + C(16x16,f32)
  return __builtin_amdgcn_wmma_f32_16x16x32_f16(false, a, false, b, (short)0, c,
                                                false, false);
}

// -------------------------------------------------------------------------
// Embedding gather: xt[(t*B+b)*E + e] = emb[review[b,t]*E + e]
// -------------------------------------------------------------------------
__global__ void embed_kernel(const int* __restrict__ review,
                             const float* __restrict__ emb,
                             float* __restrict__ xt) {
  long long i = (long long)blockIdx.x * blockDim.x + threadIdx.x;
  const long long total = (long long)S_ * B_ * E_;
  if (i >= total) return;
  int e = (int)(i % E_);
  long long r = i / E_;
  int t = (int)(r / B_);
  int b = (int)(r % B_);
  int tok = review[b * S_ + t];
  xt[i] = emb[(long long)tok * E_ + e];
}

// generic f32 -> f16 conversion
__global__ void cvt_f32_f16(const float* __restrict__ in,
                            _Float16* __restrict__ out, long long n) {
  long long i = (long long)blockIdx.x * blockDim.x + threadIdx.x;
  if (i < n) out[i] = (_Float16)in[i];
}

// -------------------------------------------------------------------------
// fp32 WMMA GEMM (precision-critical GRU input projections).
//   TRANSB = true : C[m,n] = sum_k A[m,k] * W[n,k]   (W row-major N x K)
//   TRANSB = false: C[m,n] = sum_k A[m,k] * Bm[k,n]  (Bm row-major K x N)
// Block = 256 threads computes a 64x64 tile of C. M,N multiples of 64.
// K arbitrary (zero-fill staging pads to 16). Optional per-column bias.
// -------------------------------------------------------------------------
template <bool TRANSB>
__global__ __launch_bounds__(256) void wmma_gemm_f32(
    const float* __restrict__ A, const float* __restrict__ Bm,
    const float* __restrict__ bias, float* __restrict__ C, int M, int N, int K,
    int lda, int ldb, int ldc, long long sA, long long sB, long long sC) {
  __shared__ float As[64 * 17];  // As[m][k]
  __shared__ float Bs[16 * 65];  // Bs[k][n]
  const int bz = blockIdx.z;
  A += (long long)bz * sA;
  Bm += (long long)bz * sB;
  C += (long long)bz * sC;
  const int n0 = blockIdx.x * 64;
  const int m0 = blockIdx.y * 64;
  const int tid = threadIdx.x;
  const int lane = tid & 31;
  const int wave = tid >> 5;
  const int half = lane >> 4;
  const int l16 = lane & 15;
  const int mt = wave & 3;          // M sub-tile 0..3
  const int ntA = (wave >> 2) * 2;  // N sub-tile pair base (0 or 2)
  v8f acc0 = {};
  v8f acc1 = {};
  for (int k0 = 0; k0 < K; k0 += 16) {
    {  // stage A 64x16
      int m = tid >> 2;
      int kq = (tid & 3) * 4;
      const float* arow = A + (long long)(m0 + m) * lda + k0 + kq;
#pragma unroll
      for (int i = 0; i < 4; i++) {
        int kg = k0 + kq + i;
        As[m * 17 + kq + i] = (kg < K) ? arow[i] : 0.0f;
      }
    }
    {  // stage B 16x64 as Bs[k][n]
      int n = tid & 63;
      int kq = (tid >> 6) * 4;
#pragma unroll
      for (int i = 0; i < 4; i++) {
        int kg = k0 + kq + i;
        float v = 0.0f;
        if (kg < K) {
          if (TRANSB)
            v = Bm[(long long)(n0 + n) * ldb + kg];
          else
            v = Bm[(long long)kg * ldb + (n0 + n)];
        }
        Bs[(kq + i) * 65 + n] = v;
      }
    }
    __syncthreads();
#pragma unroll
    for (int k4 = 0; k4 < 16; k4 += 4) {
      v2f a;
      a.x = As[(mt * 16 + l16) * 17 + k4 + 2 * half];
      a.y = As[(mt * 16 + l16) * 17 + k4 + 2 * half + 1];
      v2f b0, b1;
      b0.x = Bs[(k4 + 2 * half) * 65 + ntA * 16 + l16];
      b0.y = Bs[(k4 + 2 * half + 1) * 65 + ntA * 16 + l16];
      b1.x = Bs[(k4 + 2 * half) * 65 + (ntA + 1) * 16 + l16];
      b1.y = Bs[(k4 + 2 * half + 1) * 65 + (ntA + 1) * 16 + l16];
      acc0 = wmma4(a, b0, acc0);
      acc1 = wmma4(a, b1, acc1);
    }
    __syncthreads();
  }
  const int c0 = n0 + ntA * 16 + l16;
  const int c1 = c0 + 16;
  const float bb0 = bias ? bias[c0] : 0.0f;
  const float bb1 = bias ? bias[c1] : 0.0f;
#pragma unroll
  for (int j = 0; j < 8; j++) {
    int r = m0 + mt * 16 + j + 8 * half;
    C[(long long)r * ldc + c0] = acc0[j] + bb0;
    C[(long long)r * ldc + c1] = acc1[j] + bb1;
  }
}

// -------------------------------------------------------------------------
// f16 WMMA GEMM (attention block: 8x MACs/instruction vs f32 path, f32 accum).
// LDS panels stored FRAGMENT-MAJOR (row m/n: 32 contiguous k-halves, stride
// 48 halves = 96B, 16B-aligned rows). A panel, and the B panel in TRANSB
// mode, are contiguous 16B-per-lane chunks -> staged with CDNA5
// GLOBAL_LOAD_ASYNC_TO_LDS_B128 (ASYNCcnt DMA, no VGPR staging), completed
// with s_wait_asynccnt 0 before the block barrier. Non-TRANSB B panel is a
// strided column gather and stays synchronous.
// K multiple of 32; M,N multiples of 64. OUTH selects f16 vs f32 output.
// -------------------------------------------------------------------------
template <bool TRANSB, bool OUTH>
__global__ __launch_bounds__(256) void wmma_gemm_f16(
    const _Float16* __restrict__ A, const _Float16* __restrict__ Bm,
    void* __restrict__ Cv, int M, int N, int K, int lda, int ldb, int ldc,
    long long sA, long long sB, long long sC) {
  __shared__ _Float16 As[64 * 48];  // As[m][k0..31], stride 48
  __shared__ _Float16 Bs[64 * 48];  // Bs[n][k0..31], stride 48
  const int bz = blockIdx.z;
  A += (long long)bz * sA;
  Bm += (long long)bz * sB;
  const int n0 = blockIdx.x * 64;
  const int m0 = blockIdx.y * 64;
  const int tid = threadIdx.x;
  const int lane = tid & 31;
  const int wave = tid >> 5;
  const int hf = lane >> 4;
  const int l16 = lane & 15;
  const int mt = wave & 3;
  const int ntA = (wave >> 2) * 2;
  v8f acc0 = {};
  v8f acc1 = {};
  for (int k0 = 0; k0 < K; k0 += 32) {
    {  // stage A: 64 rows x 32 halves, one 16B chunk per thread
      int m = tid >> 2;
      int kq = (tid & 3) * 8;
      const _Float16* arow = A + (long long)(m0 + m) * lda + k0 + kq;
      _Float16* adst = As + m * 48 + kq;
#if ASYNC_LDS
      __builtin_amdgcn_global_load_async_to_lds_b128(ASYNC_PTR(arow),
                                                     ASYNC_PTR(adst), 0, 0);
#else
#pragma unroll
      for (int i = 0; i < 8; i++) adst[i] = arow[i];
#endif
    }
    {  // stage B fragment-major: Bs[n][k]
      int n = tid & 63;
      int kq = (tid >> 6) * 8;
      _Float16* bdst = Bs + n * 48 + kq;
      if (TRANSB) {  // contiguous along k in memory
        const _Float16* brow = Bm + (long long)(n0 + n) * ldb + k0 + kq;
#if ASYNC_LDS
        __builtin_amdgcn_global_load_async_to_lds_b128(ASYNC_PTR(brow),
                                                       ASYNC_PTR(bdst), 0, 0);
#else
#pragma unroll
        for (int i = 0; i < 8; i++) bdst[i] = brow[i];
#endif
      } else {  // strided column gather, synchronous
#pragma unroll
        for (int i = 0; i < 8; i++)
          bdst[i] = Bm[(long long)(k0 + kq + i) * ldb + (n0 + n)];
      }
    }
#if ASYNC_LDS
    __builtin_amdgcn_s_wait_asynccnt(0);
#endif
    __syncthreads();
    // A fragment: lane (hf,l16) holds k = {hf*8+0..7, 16+hf*8+0..7}
    v16h a;
    {
      const _Float16* ar = As + (mt * 16 + l16) * 48;
      v8h alo = *(const v8h*)(ar + hf * 8);
      v8h ahi = *(const v8h*)(ar + 16 + hf * 8);
#pragma unroll
      for (int h = 0; h < 8; h++) {
        a[h] = alo[h];
        a[h + 8] = ahi[h];
      }
    }
    // B fragments: lane (hf,l16) holds k = {hf*8+0..7, 16+hf*8+0..7}
    v16h b0, b1;
    {
      const _Float16* br0 = Bs + (ntA * 16 + l16) * 48 + hf * 8;
      const _Float16* br1 = Bs + ((ntA + 1) * 16 + l16) * 48 + hf * 8;
      v8h p0 = *(const v8h*)(br0);
      v8h p1 = *(const v8h*)(br0 + 16);
      v8h q0 = *(const v8h*)(br1);
      v8h q1 = *(const v8h*)(br1 + 16);
#pragma unroll
      for (int h = 0; h < 8; h++) {
        b0[h] = p0[h];
        b0[h + 8] = p1[h];
        b1[h] = q0[h];
        b1[h + 8] = q1[h];
      }
    }
    acc0 = wmma32h(a, b0, acc0);
    acc1 = wmma32h(a, b1, acc1);
    __syncthreads();
  }
  const int c0 = n0 + ntA * 16 + l16;
  const int c1 = c0 + 16;
  if (OUTH) {
    _Float16* C = (_Float16*)Cv + (long long)bz * sC;
#pragma unroll
    for (int j = 0; j < 8; j++) {
      int r = m0 + mt * 16 + j + 8 * hf;
      C[(long long)r * ldc + c0] = (_Float16)acc0[j];
      C[(long long)r * ldc + c1] = (_Float16)acc1[j];
    }
  } else {
    float* C = (float*)Cv + (long long)bz * sC;
#pragma unroll
    for (int j = 0; j < 8; j++) {
      int r = m0 + mt * 16 + j + 8 * hf;
      C[(long long)r * ldc + c0] = acc0[j];
      C[(long long)r * ldc + c1] = acc1[j];
    }
  }
}

// -------------------------------------------------------------------------
// GRU recurrence (fp32 WMMA), one direction per block.
// 1024 threads = 32 waves; wave owns the same (batch-tile, col-tile) across
// all three gates so gate math stays in registers; h (32x256) in LDS.
// -------------------------------------------------------------------------
__global__ __launch_bounds__(1024) void gru_layer_kernel(
    const float* __restrict__ giF, const float* __restrict__ giB,
    const float* __restrict__ WhhF, const float* __restrict__ WhhB,
    const float* __restrict__ bhhF, const float* __restrict__ bhhB,
    float* __restrict__ out, int outT, int outB) {
  __shared__ float hbuf[32 * 257];
  const int dir = blockIdx.x;
  const float* gi = dir ? giB : giF;
  const float* Whh = dir ? WhhB : WhhF;
  const float* bhh = dir ? bhhB : bhhF;
  const int dirOff = dir * H_;
  const int tid = threadIdx.x;
  for (int i = tid; i < 32 * 257; i += 1024) hbuf[i] = 0.0f;
  __syncthreads();
  const int lane = tid & 31;
  const int wave = tid >> 5;
  const int half = lane >> 4;
  const int l16 = lane & 15;
  const int mt = wave & 1;   // batch tile (rows 0..15 / 16..31)
  const int ct = wave >> 1;  // column tile 0..15 within each gate
  const int c = ct * 16 + l16;
  const float br_ = bhh[c];
  const float bz_ = bhh[H_ + c];
  const float bn_ = bhh[2 * H_ + c];
  const float* wr = Whh + (long long)(0 * H_ + c) * H_;
  const float* wz = Whh + (long long)(1 * H_ + c) * H_;
  const float* wn = Whh + (long long)(2 * H_ + c) * H_;
  for (int step = 0; step < S_; step++) {
    const int t = dir ? (S_ - 1 - step) : step;
    const long long gibase = (long long)t * B_ * H3_;
    __builtin_prefetch(gi + gibase, 0, 0);  // global_prefetch of step's gates
    v8f ar = {};
    v8f az = {};
    v8f an = {};
#pragma unroll 4
    for (int k4 = 0; k4 < H_; k4 += 4) {
      const int kk = k4 + 2 * half;
      v2f a;
      a.x = hbuf[(mt * 16 + l16) * 257 + kk];
      a.y = hbuf[(mt * 16 + l16) * 257 + kk + 1];
      v2f brv, bzv, bnv;
      brv.x = wr[kk];
      brv.y = wr[kk + 1];
      bzv.x = wz[kk];
      bzv.y = wz[kk + 1];
      bnv.x = wn[kk];
      bnv.y = wn[kk + 1];
      ar = wmma4(a, brv, ar);
      az = wmma4(a, bzv, az);
      an = wmma4(a, bnv, an);
    }
    __syncthreads();  // all waves done reading h before any update
#pragma unroll
    for (int j = 0; j < 8; j++) {
      const int b = mt * 16 + j + 8 * half;
      const long long gir = gibase + (long long)b * H3_;
      const float gr = gi[gir + c] + ar[j] + br_;
      const float gz = gi[gir + H_ + c] + az[j] + bz_;
      const float gni = gi[gir + 2 * H_ + c];
      const float r = 1.0f / (1.0f + __expf(-gr));
      const float z = 1.0f / (1.0f + __expf(-gz));
      const float n = tanhf(gni + r * (an[j] + bn_));
      const float hprev = hbuf[b * 257 + c];
      const float hnew = (1.0f - z) * n + z * hprev;
      hbuf[b * 257 + c] = hnew;
      out[(long long)t * outT + (long long)b * outB + dirOff + c] = hnew;
    }
    __syncthreads();
  }
}

// final[b,0:256] = review_h[b,S-1,0:256]; final[b,256:512] = review_h[b,0,256:512]
__global__ void final_gather(const float* __restrict__ rvh,
                             float* __restrict__ fin) {
  int i = blockIdx.x * 256 + threadIdx.x;
  if (i >= B_ * H2_) return;
  int b = i >> 9;
  int cc = i & 511;
  long long base = (long long)b * S_ * H2_;
  fin[i] =
      (cc < H_) ? rvh[base + (long long)(S_ - 1) * H2_ + cc] : rvh[base + cc];
}

// scores[b,t] = <proj[b,t,:], final[b,:]> + (mask-1)*NEG ; softmax over t
__global__ __launch_bounds__(512) void scores_attn(
    const float* __restrict__ proj, const float* __restrict__ fin,
    const float* __restrict__ mask, float* __restrict__ attn) {
  __shared__ float red[512];
  const int b = blockIdx.x;
  const int t = threadIdx.x;
  const float* p = proj + (long long)b * S_ * H2_ + (long long)t * H2_;
  const float* f = fin + b * H2_;
  float s = 0.0f;
  for (int k = 0; k < H2_; k++) s += p[k] * f[k];
  s += (mask[b * S_ + t] - 1.0f) * NEG_;
  red[t] = s;
  __syncthreads();
  for (int off = 256; off; off >>= 1) {
    if (t < off) red[t] = fmaxf(red[t], red[t + off]);
    __syncthreads();
  }
  const float mx = red[0];
  __syncthreads();
  const float e = __expf(s - mx);
  red[t] = e;
  __syncthreads();
  for (int off = 256; off; off >>= 1) {
    if (t < off) red[t] += red[t + off];
    __syncthreads();
  }
  attn[b * S_ + t] = e / red[0];
}

// gc[b,d] = sum_t attn[b,t] * review_h[b,t,d]
__global__ __launch_bounds__(512) void gc_kernel(const float* __restrict__ attn,
                                                 const float* __restrict__ rvh,
                                                 float* __restrict__ gc) {
  const int b = blockIdx.x;
  const int d = threadIdx.x;
  const float* a = attn + b * S_;
  const float* r = rvh + (long long)b * S_ * H2_ + d;
  float s = 0.0f;
  for (int t = 0; t < S_; t++) s += a[t] * r[(long long)t * H2_];
  gc[b * H2_ + d] = s;
}

// rh_h[b,t,0:512]=review_h[b,t,:]; rh_h[b,t,512:1024]=gc[b,:]  (f16 output)
__global__ void build_rh(const float* __restrict__ rvh,
                         const float* __restrict__ gc,
                         _Float16* __restrict__ rh) {
  long long i = (long long)blockIdx.x * 256 + threadIdx.x;
  const long long total = (long long)B_ * S_ * H4_;
  if (i >= total) return;
  int cc = (int)(i & 1023);
  long long row = i >> 10;
  int b = (int)(row >> 9);
  int t = (int)(row & 511);
  float v = (cc < H2_)
                ? rvh[(long long)b * S_ * H2_ + (long long)t * H2_ + cc]
                : gc[b * H2_ + (cc - H2_)];
  rh[i] = (_Float16)v;
}

// softmax over key axis (len 512) with +bias_m and key mask; f32 in, f16 out.
__global__ __launch_bounds__(256) void alpha_softmax(
    const float* __restrict__ alpha, _Float16* __restrict__ alpha_h,
    const float* __restrict__ mask, const float* __restrict__ bias_m) {
  __shared__ float red[256];
  const int row = blockIdx.x;  // b*S + s
  const int b = row >> 9;
  const float* a = alpha + (long long)row * S_;
  _Float16* ah = alpha_h + (long long)row * S_;
  const int tid = threadIdx.x;
  const float bm = bias_m[0];
  float v0 = a[tid] + bm + (mask[b * S_ + tid] - 1.0f) * NEG_;
  float v1 = a[tid + 256] + bm + (mask[b * S_ + tid + 256] - 1.0f) * NEG_;
  red[tid] = fmaxf(v0, v1);
  __syncthreads();
  for (int off = 128; off; off >>= 1) {
    if (tid < off) red[tid] = fmaxf(red[tid], red[tid + off]);
    __syncthreads();
  }
  const float mx = red[0];
  __syncthreads();
  const float e0 = __expf(v0 - mx);
  const float e1 = __expf(v1 - mx);
  red[tid] = e0 + e1;
  __syncthreads();
  for (int off = 128; off; off >>= 1) {
    if (tid < off) red[tid] += red[tid + off];
    __syncthreads();
  }
  const float inv = 1.0f / red[0];
  ah[tid] = (_Float16)(e0 * inv);
  ah[tid + 256] = (_Float16)(e1 * inv);
}

// logits = s_i @ w_r^T (OUT=3), log_softmax, * mask
__global__ void logits_kernel(const float* __restrict__ s_i,
                              const float* __restrict__ w_r,
                              const float* __restrict__ mask,
                              float* __restrict__ out) {
  const int idx = blockIdx.x * 256 + threadIdx.x;
  if (idx >= B_ * S_) return;
  const float* srow = s_i + (long long)idx * H4_;
  float l[3];
#pragma unroll
  for (int o = 0; o < 3; o++) {
    const float* w = w_r + o * H4_;
    float s = 0.0f;
    for (int k = 0; k < H4_; k++) s += srow[k] * w[k];
    l[o] = s;
  }
  const float mx = fmaxf(l[0], fmaxf(l[1], l[2]));
  const float z = __expf(l[0] - mx) + __expf(l[1] - mx) + __expf(l[2] - mx);
  const float lz = __logf(z) + mx;
  const float m = mask[idx];
#pragma unroll
  for (int o = 0; o < 3; o++) out[idx * 3 + o] = (l[o] - lz) * m;
}

// -------------------------------------------------------------------------
// Host side: workspace arena with lifetime reuse (peak ~243 MB).
// -------------------------------------------------------------------------
extern "C" void kernel_launch(void* const* d_in, const int* in_sizes, int n_in,
                              void* d_out, int out_size, void* d_ws,
                              size_t ws_size, hipStream_t stream) {
  const int* review = (const int*)d_in[0];
  const float* mask = (const float*)d_in[1];
  const float* emb = (const float*)d_in[2];
  const float* w_a = (const float*)d_in[3];
  const float* weight_m = (const float*)d_in[4];
  const float* bias_m = (const float*)d_in[5];
  const float* w_r = (const float*)d_in[6];
  const float* Wih0f = (const float*)d_in[7];
  const float* Whh0f = (const float*)d_in[8];
  const float* bih0f = (const float*)d_in[9];
  const float* bhh0f = (const float*)d_in[10];
  const float* Wih0b = (const float*)d_in[11];
  const float* Whh0b = (const float*)d_in[12];
  const float* bih0b = (const float*)d_in[13];
  const float* bhh0b = (const float*)d_in[14];
  const float* Wih1f = (const float*)d_in[15];
  const float* Whh1f = (const float*)d_in[16];
  const float* bih1f = (const float*)d_in[17];
  const float* bhh1f = (const float*)d_in[18];
  const float* Wih1b = (const float*)d_in[19];
  const float* Whh1b = (const float*)d_in[20];
  const float* bih1b = (const float*)d_in[21];
  const float* bhh1b = (const float*)d_in[22];
  float* out = (float*)d_out;

  char* ws = (char*)d_ws;
  // arena offsets (bytes); lifetimes allow reuse
  const long long o_xt = 0;                    // xt 19,660,800 ; later small
  const long long o_giF = 19660800;            // gi slab A (50,331,648)
  const long long o_giB = o_giF + 50331648LL;  // gi slab B (50,331,648)
  const long long o_l0 = o_giB + 50331648LL;   // l0 33,554,432 ; later proj
  const long long o_rvh = o_l0 + 33554432LL;   // rvh 33,554,432 ; later alpha
  const long long o_g2 = o_rvh + 33554432LL;   // s_i f32 (67,108,864)
  float* xt = (float*)(ws + o_xt);
  float* giF = (float*)(ws + o_giF);
  float* giB = (float*)(ws + o_giB);
  float* l0 = (float*)(ws + o_l0);
  float* rvh = (float*)(ws + o_rvh);
  float* s_i = (float*)(ws + o_g2);
  // gi slabs reused after layer-1 recurrence for the f16 attention tensors:
  _Float16* rvh_h = (_Float16*)(ws + o_giF);                 // 16,777,216
  _Float16* rh_h = (_Float16*)(ws + o_giF + 16777216LL);     // 33,554,432
  _Float16* rhM_h = (_Float16*)(ws + o_giF + 50331648LL);    // 33,554,432
  _Float16* alpha_h = (_Float16*)(ws + o_giF + 83886080LL);  // 16,777,216
  float* proj = (float*)(ws + o_l0);    // reuses l0 slab
  float* alpha = (float*)(ws + o_rvh);  // reuses rvh slab (rvh dead by then)
  float* attn = (float*)(ws + o_xt);    // xt slab reused for small buffers
  float* gc = (float*)(ws + o_xt + 65536);
  float* fin = (float*)(ws + o_xt + 131072);
  _Float16* wa_h = (_Float16*)(ws + o_xt + 196608);  // 524,288
  _Float16* wm_h = (_Float16*)(ws + o_xt + 720896);  // 2,097,152

  // 1. embedding gather -> xt (S,B,E)
  {
    long long total = (long long)S_ * B_ * E_;
    embed_kernel<<<(unsigned)((total + 255) / 256), 256, 0, stream>>>(review,
                                                                      emb, xt);
  }
  // 2. layer-0 input projections gi = xt @ Wih^T + bih (fp32 WMMA, K=300)
  wmma_gemm_f32<true><<<dim3(H3_ / 64, (S_ * B_) / 64, 1), 256, 0, stream>>>(
      xt, Wih0f, bih0f, giF, S_ * B_, H3_, E_, E_, E_, H3_, 0, 0, 0);
  wmma_gemm_f32<true><<<dim3(H3_ / 64, (S_ * B_) / 64, 1), 256, 0, stream>>>(
      xt, Wih0b, bih0b, giB, S_ * B_, H3_, E_, E_, E_, H3_, 0, 0, 0);
  // 3. layer-0 recurrence -> l0 (row t*B+b, concat layout)
  gru_layer_kernel<<<2, 1024, 0, stream>>>(giF, giB, Whh0f, Whh0b, bhh0f,
                                           bhh0b, l0, B_ * H2_, H2_);
  // 4. layer-1 input projections (K=512)
  wmma_gemm_f32<true><<<dim3(H3_ / 64, (S_ * B_) / 64, 1), 256, 0, stream>>>(
      l0, Wih1f, bih1f, giF, S_ * B_, H3_, H2_, H2_, H2_, H3_, 0, 0, 0);
  wmma_gemm_f32<true><<<dim3(H3_ / 64, (S_ * B_) / 64, 1), 256, 0, stream>>>(
      l0, Wih1b, bih1b, giB, S_ * B_, H3_, H2_, H2_, H2_, H3_, 0, 0, 0);
  // 5. layer-1 recurrence -> review_h (B,S,2H)
  gru_layer_kernel<<<2, 1024, 0, stream>>>(giF, giB, Whh1f, Whh1b, bhh1f,
                                           bhh1b, rvh, H2_, S_ * H2_);
  // 6. final hidden (B,2H)
  final_gather<<<(B_ * H2_ + 255) / 256, 256, 0, stream>>>(rvh, fin);
  // 7. f16 copies for the attention block
  cvt_f32_f16<<<(unsigned)(((long long)B_ * S_ * H2_ + 255) / 256), 256, 0,
                stream>>>(rvh, rvh_h, (long long)B_ * S_ * H2_);
  cvt_f32_f16<<<(H2_ * H2_ + 255) / 256, 256, 0, stream>>>(
      w_a, wa_h, (long long)H2_ * H2_);
  cvt_f32_f16<<<(H4_ * H4_ + 255) / 256, 256, 0, stream>>>(
      weight_m, wm_h, (long long)H4_ * H4_);
  // 8. proj = review_h @ w_a^T (f16 WMMA, f32 out), batched over B
  wmma_gemm_f16<true, false><<<dim3(H2_ / 64, S_ / 64, B_), 256, 0, stream>>>(
      rvh_h, wa_h, proj, S_, H2_, H2_, H2_, H2_, H2_, (long long)S_ * H2_, 0,
      (long long)S_ * H2_);
  // 9. attention scores + softmax; 10. global context
  scores_attn<<<B_, 512, 0, stream>>>(proj, fin, mask, attn);
  gc_kernel<<<B_, 512, 0, stream>>>(attn, rvh, gc);
  // 11. rh = concat(review_h, gc) -> f16
  {
    long long total = (long long)B_ * S_ * H4_;
    build_rh<<<(unsigned)((total + 255) / 256), 256, 0, stream>>>(rvh, gc,
                                                                  rh_h);
  }
  // 12. rhM = rh @ weight_m (f16 WMMA, f16 out), 16384x1024x1024
  wmma_gemm_f16<false, true>
      <<<dim3(H4_ / 64, (B_ * S_) / 64, 1), 256, 0, stream>>>(
          rh_h, wm_h, rhM_h, B_ * S_, H4_, H4_, H4_, H4_, H4_, 0, 0, 0);
  // 13. alpha = rhM @ rh^T per batch (512x512x1024), f32 out for softmax
  wmma_gemm_f16<true, false><<<dim3(S_ / 64, S_ / 64, B_), 256, 0, stream>>>(
      rhM_h, rh_h, alpha, S_, S_, H4_, H4_, H4_, S_, (long long)S_ * H4_,
      (long long)S_ * H4_, (long long)S_ * S_);
  // 14. softmax over keys (+bias_m, key mask) -> f16
  alpha_softmax<<<B_ * S_, 256, 0, stream>>>(alpha, alpha_h, mask, bias_m);
  // 15. s_i = alpha @ rh per batch (512x1024x512), f32 out
  wmma_gemm_f16<false, false><<<dim3(H4_ / 64, S_ / 64, B_), 256, 0, stream>>>(
      alpha_h, rh_h, s_i, S_, H4_, S_, S_, H4_, H4_, (long long)S_ * S_,
      (long long)S_ * H4_, (long long)S_ * H4_);
  // 16. logits + log_softmax * mask -> out
  logits_kernel<<<(B_ * S_ + 255) / 256, 256, 0, stream>>>(s_i, w_r, mask, out);
}
